// MaskedMultiHeadAttention_76897094468067
// MI455X (gfx1250) — matrix operand order repaired and checked by reading, not compiled, not run on hardware
//
#include <hip/hip_runtime.h>
#include <hip/hip_bf16.h>

#define B_   4
#define N_   2048
#define D_   1024
#define H_   16
#define DK_  64
#define DV_  64
#define HDV_ 1024

#define DCH  128           // K-chunk staged in LDS per step
#define SP   (DCH + 8)     // padded LDS row stride (+16B) -> conflict-free
#define NST  (D_ / DCH)    // 8 stages

typedef __attribute__((ext_vector_type(16))) _Float16 v16h;
typedef __attribute__((ext_vector_type(8)))  _Float16 v8h;
typedef __attribute__((ext_vector_type(8)))  float    v8f;
typedef __attribute__((ext_vector_type(4)))  float    v4f;
typedef __attribute__((ext_vector_type(4)))  unsigned int u32x4;
typedef __attribute__((ext_vector_type(8)))  int      i32x8;
typedef __attribute__((ext_vector_type(4)))  int      i32x4;

#if defined(__gfx1250__) && __has_builtin(__builtin_amdgcn_tensor_load_to_lds) && \
    __has_builtin(__builtin_amdgcn_s_wait_tensorcnt)
#define HAS_TDM 1
#else
#define HAS_TDM 0
#endif

// D = A(16x32 f16) * B(32x16 f16) + C(16x16 f32)
__device__ __forceinline__ v8f wmma16(v16h a, v16h b, v8f c) {
  return __builtin_amdgcn_wmma_f32_16x16x32_f16(
      false, a, false, b, (short)0, c, false, false);
}

// B-fragment: elements 0..7 = p[0..7], 8..15 = p[16..23]
__device__ __forceinline__ v16h ldB(const _Float16* p) {
  v8h a = *(const v8h*)p;
  v8h b = *(const v8h*)(p + 16);
  v16h r;
#pragma unroll
  for (int i = 0; i < 8; ++i) { r[i] = a[i]; r[8 + i] = b[i]; }
  return r;
}

// A-fragment from f16: elements 0..7 = p[0..7], 8..15 = p[16..23]
__device__ __forceinline__ v16h ldA16(const _Float16* p) {
  v8h a = *(const v8h*)p;
  v8h b = *(const v8h*)(p + 16);
  v16h r;
#pragma unroll
  for (int i = 0; i < 8; ++i) { r[i] = a[i]; r[8 + i] = b[i]; }
  return r;
}

// A-fragment from f32 source (convert in-register)
__device__ __forceinline__ v16h ldA32(const float* p) {
  v4f a0 = *(const v4f*)(p);
  v4f a1 = *(const v4f*)(p + 4);
  v4f b0 = *(const v4f*)(p + 16);
  v4f b1 = *(const v4f*)(p + 20);
  v16h r;
#pragma unroll
  for (int i = 0; i < 4; ++i) {
    r[i]      = (_Float16)a0[i];
    r[4 + i]  = (_Float16)a1[i];
    r[8 + i]  = (_Float16)b0[i];
    r[12 + i] = (_Float16)b1[i];
  }
  return r;
}

// ---------------------------------------------------------------------------
// Stage a 64-row x DCH-col f16 tile (row stride gstride elements in global)
// into LDS with +8-half row padding. TDM path: one DMA issued by wave 0,
// D# pad fields reproduce the padded layout (pad_interval=5 -> every 64 DW,
// pad_amount=3 -> 4 DW). Fallback: cooperative b128 copy.
// ---------------------------------------------------------------------------
__device__ __forceinline__ void stage_tile(_Float16* sdst, const _Float16* gsrc,
                                           int gstride) {
#if HAS_TDM
  if (threadIdx.x < 32) {
    const unsigned long long ga = (unsigned long long)(uintptr_t)gsrc;
    u32x4 g0;
    g0[0] = 1u;                                        // count=1, flags 0
    g0[1] = (unsigned int)(uintptr_t)sdst;             // LDS byte offset
    g0[2] = (unsigned int)ga;                          // global addr lo
    g0[3] = (unsigned int)(ga >> 32) | (2u << 30);     // addr hi | type=2
    i32x8 g1;
    g1[0] = (int)((1u << 16)           // data_size = 2 bytes
                | (1u << 20)           // pad_enable
                | (5u << 22)           // pad_interval: 64 DWORDs
                | (3u << 25));         // pad_amount: 4 DWORDs (16B)
    g1[1] = (int)((unsigned)DCH << 16);                      // tensor_dim0 lo16
    g1[2] = (int)(((unsigned)DCH >> 16) | ((unsigned)DK_ << 16)); // dim0 hi | dim1 lo
    g1[3] = (int)(((unsigned)DK_ >> 16) | ((unsigned)DCH << 16)); // dim1 hi | tile0
    g1[4] = (int)DK_;                                        // tile_dim1 (tile2=0)
    g1[5] = (int)gstride;                                    // dim0 stride lo32
    g1[6] = 0;
    g1[7] = 0;
    i32x4 z4 = {0, 0, 0, 0};
#if __clang_major__ >= 23
    i32x8 z8 = {0, 0, 0, 0, 0, 0, 0, 0};
    __builtin_amdgcn_tensor_load_to_lds(g0, g1, z4, z4, z8, 0);
#else
    __builtin_amdgcn_tensor_load_to_lds(g0, g1, z4, z4, 0);
#endif
  }
#else
  for (int idx = threadIdx.x; idx < DK_ * (DCH / 8); idx += 128) {
    const int kk = idx / (DCH / 8);
    const int j  = (idx % (DCH / 8)) * 8;
    *(v8h*)&sdst[kk * SP + j] = *(const v8h*)(gsrc + (size_t)kk * gstride + j);
  }
#endif
}

__device__ __forceinline__ void stage_wait(bool more) {
#if HAS_TDM
  if (threadIdx.x < 32) {
    if (more) __builtin_amdgcn_s_wait_tensorcnt(1);
    else      __builtin_amdgcn_s_wait_tensorcnt(0);
  }
#else
  (void)more;
#endif
  __syncthreads();
}

// ---------------------------------------------------------------------------
// Kernel 1: weight transpose + f32->f16 convert.
// wq/wk/wv [H,D,64] -> wT [3][H][64][D];  wo [1024,1024] -> woT [d][k]
// ---------------------------------------------------------------------------
__global__ void prep_kernel(const float* __restrict__ wq, const float* __restrict__ wk,
                            const float* __restrict__ wv, const float* __restrict__ wo,
                            _Float16* __restrict__ wT, _Float16* __restrict__ woT) {
  int i = blockIdx.x * 256 + threadIdx.x;
  const int WSZ = 3 * H_ * DK_ * D_;
  if (i < WSZ) {
    int t   = i / (H_ * DK_ * D_);
    int rem = i % (H_ * DK_ * D_);
    int h   = rem / (DK_ * D_);
    int r2  = rem % (DK_ * D_);
    int k   = r2 / D_;
    int d   = r2 % D_;
    const float* w = (t == 0) ? wq : (t == 1) ? wk : wv;
    wT[i] = (_Float16)w[((size_t)h * D_ + d) * DK_ + k];
  } else {
    int j = i - WSZ;
    int d = j / HDV_;
    int k = j % HDV_;
    woT[j] = (_Float16)wo[(size_t)k * D_ + d];
  }
}

// ---------------------------------------------------------------------------
// Kernel 2: per-head QKV projection. Block = 4 waves, 128 rows x 64 cols.
// Weight tile TDM-staged into padded LDS (double-buffered); each wave does a
// 32x64 tile -> every LDS B-fragment feeds 2 WMMAs.
// grid: (N/128, 3*H, B)
// ---------------------------------------------------------------------------
__global__ __launch_bounds__(128) void proj_kernel(
    const float* __restrict__ q, const float* __restrict__ k, const float* __restrict__ v,
    const float* __restrict__ bq, const float* __restrict__ bk, const float* __restrict__ bv,
    const _Float16* __restrict__ wT,
    _Float16* __restrict__ qh, _Float16* __restrict__ kh, _Float16* __restrict__ vhT) {
  __shared__ _Float16 sW[2][DK_ * SP];
  const int lane = threadIdx.x & 31;
  const int wave = threadIdx.x >> 5;
  const int b = blockIdx.z;
  const int t = blockIdx.y / H_;
  const int h = blockIdx.y % H_;
  const int l15 = lane & 15;
  const int hi  = (lane >> 4) & 1;
  const int n0  = blockIdx.x * 128 + wave * 32;

  const float* X = ((t == 0) ? q : (t == 1) ? k : v) + (size_t)b * N_ * D_;
  const _Float16* W = wT + (size_t)(t * H_ + h) * DK_ * D_;
  const float* bias = ((t == 0) ? bq : (t == 1) ? bk : bv) + h * DK_;

  v8f acc[2][4] = {};
  const float* arow0 = X + (size_t)(n0 + l15) * D_;
  const float* arow1 = arow0 + (size_t)16 * D_;

  stage_tile(sW[0], W, D_);
  for (int s = 0; s < NST; ++s) {
    if (s + 1 < NST) stage_tile(sW[(s + 1) & 1], W + (s + 1) * DCH, D_);
    stage_wait(s + 1 < NST);
    const _Float16* sb = sW[s & 1];
#pragma unroll
    for (int dd = 0; dd < DCH; dd += 32) {
      const int d0 = s * DCH + dd;
      v16h A0 = ldA32(arow0 + d0 + hi * 8);
      v16h A1 = ldA32(arow1 + d0 + hi * 8);
#pragma unroll
      for (int ct = 0; ct < 4; ++ct) {
        v16h Bv = ldB(sb + (ct * 16 + l15) * SP + dd + hi * 8);
        acc[0][ct] = wmma16(A0, Bv, acc[0][ct]);
        acc[1][ct] = wmma16(A1, Bv, acc[1][ct]);
      }
    }
    __syncthreads();
  }

  if (t < 2) {
    _Float16* O = ((t == 0) ? qh : kh) + (size_t)(b * H_ + h) * N_ * DK_;
#pragma unroll
    for (int rg = 0; rg < 2; ++rg) {
      const int rbase = n0 + rg * 16 + hi * 8;
#pragma unroll
      for (int ct = 0; ct < 4; ++ct) {
        const int col = ct * 16 + l15;
        const float bia = bias[col];
#pragma unroll
        for (int r = 0; r < 8; ++r)
          O[(size_t)(rbase + r) * DK_ + col] = (_Float16)(acc[rg][ct][r] + bia);
      }
    }
  } else {
    _Float16* O = vhT + (size_t)(b * H_ + h) * DV_ * N_;
#pragma unroll
    for (int rg = 0; rg < 2; ++rg) {
      const int rbase = n0 + rg * 16 + hi * 8;
#pragma unroll
      for (int ct = 0; ct < 4; ++ct) {
        const int dv = ct * 16 + l15;
        const float bia = bias[dv];
        v8h pk;
#pragma unroll
        for (int r = 0; r < 8; ++r) pk[r] = (_Float16)(acc[rg][ct][r] + bia);
        *(v8h*)(O + (size_t)dv * N_ + rbase) = pk;
      }
    }
  }
}

// ---------------------------------------------------------------------------
// Kernel 3: flash attention. S^T = K * Q^T so each lane owns one query column
// -> lane-local online softmax (one shfl_xor(16) to the partner lane holding
// the complementary key rows). O = V^T * P, ctx f16 row-major.
// grid: (N/64, H, B); block 128; each wave = 16 queries.
// ---------------------------------------------------------------------------
__global__ __launch_bounds__(128) void attn_kernel(
    const _Float16* __restrict__ qh, const _Float16* __restrict__ kh,
    const _Float16* __restrict__ vhT, _Float16* __restrict__ ctx) {
  const int lane = threadIdx.x & 31;
  const int wave = threadIdx.x >> 5;
  const int h = blockIdx.y, b = blockIdx.z;
  const int n0 = blockIdx.x * 64 + wave * 16;
  const int l15 = lane & 15;
  const int hi  = (lane >> 4) & 1;

  const size_t bh = (size_t)(b * H_ + h);
  const _Float16* Qm = qh + bh * N_ * DK_;
  const _Float16* Km = kh + bh * N_ * DK_;
  const _Float16* Vt = vhT + bh * DV_ * N_;

  const int nq = n0 + l15;
  const _Float16* qp = Qm + (size_t)nq * DK_ + hi * 8;
  const v16h bq0 = ldB(qp);
  const v16h bq1 = ldB(qp + 32);

  v8f o[4] = {};
  float mrun = -3.0e38f, lsum = 0.0f;
  const float scale = 0.125f;

  const int nblocks = (n0 + 16 + 31) / 32;
  for (int blk = 0; blk < nblocks; ++blk) {
    const int m0 = blk * 32;
    if (blk + 1 < nblocks) {
      __builtin_prefetch(Km + (size_t)(m0 + 32 + l15) * DK_, 0, 0);
      __builtin_prefetch(Vt + (size_t)l15 * N_ + m0 + 32, 0, 0);
    }

    v8f st0 = {}, st1 = {};
    {
      const _Float16* kp0 = Km + (size_t)(m0 + l15) * DK_ + hi * 8;
      st0 = wmma16(ldA16(kp0),      bq0, st0);
      st0 = wmma16(ldA16(kp0 + 32), bq1, st0);
      const _Float16* kp1 = kp0 + 16 * DK_;
      st1 = wmma16(ldA16(kp1),      bq0, st1);
      st1 = wmma16(ldA16(kp1 + 32), bq1, st1);
    }

    const int mb = m0 + hi * 8;
    float s_lo[8], s_hi[8];
    float bmax = -3.0e38f;
#pragma unroll
    for (int r = 0; r < 8; ++r) {
      float s0 = st0[r] * scale; if (mb + r      > nq) s0 = -3.0e38f;
      float s1 = st1[r] * scale; if (mb + 16 + r > nq) s1 = -3.0e38f;
      s_lo[r] = s0; s_hi[r] = s1;
      bmax = fmaxf(bmax, fmaxf(s0, s1));
    }
    bmax = fmaxf(bmax, __shfl_xor(bmax, 16, 32));
    const float mnew  = fmaxf(mrun, bmax);
    const float alpha = __expf(mrun - mnew);
    mrun = mnew;

    float p_lo[8], p_hi[8], ls = 0.0f;
#pragma unroll
    for (int r = 0; r < 8; ++r) {
      p_lo[r] = __expf(s_lo[r] - mnew);
      p_hi[r] = __expf(s_hi[r] - mnew);
      ls += p_lo[r] + p_hi[r];
    }
    ls += __shfl_xor(ls, 16, 32);
    lsum = lsum * alpha + ls;

#pragma unroll
    for (int t = 0; t < 4; ++t)
#pragma unroll
      for (int r = 0; r < 8; ++r) o[t][r] *= alpha;

    v16h Pb;
#pragma unroll
    for (int r = 0; r < 8; ++r) {
      Pb[r]     = (_Float16)p_lo[r];
      Pb[8 + r] = (_Float16)p_hi[r];
    }

#pragma unroll
    for (int t = 0; t < 4; ++t) {
      const _Float16* vp = Vt + (size_t)(t * 16 + l15) * N_ + m0 + hi * 8;
      o[t] = wmma16(ldA16(vp), Pb, o[t]);
    }
  }

  const float inv = 1.0f / lsum;
  _Float16* C = ctx + ((size_t)b * N_ + nq) * HDV_ + h * DV_;
#pragma unroll
  for (int t = 0; t < 4; ++t) {
    v8h pk;
#pragma unroll
    for (int r = 0; r < 8; ++r) pk[r] = (_Float16)(o[t][r] * inv);
    *(v8h*)(C + t * 16 + hi * 8) = pk;
  }
}

// ---------------------------------------------------------------------------
// Kernel 4: output projection: out = ctx[8192x1024] x wo + bo (f32 out).
// Same TDM-staged LDS weight scheme; 32x64 tile per wave.
// grid: (8192/128, 1024/64)
// ---------------------------------------------------------------------------
__global__ __launch_bounds__(128) void oproj_kernel(
    const _Float16* __restrict__ ctx, const _Float16* __restrict__ woT,
    const float* __restrict__ bo, float* __restrict__ out) {
  __shared__ _Float16 sW[2][DK_ * SP];
  const int lane = threadIdx.x & 31;
  const int wave = threadIdx.x >> 5;
  const int l15 = lane & 15;
  const int hi  = (lane >> 4) & 1;
  const int r0 = blockIdx.x * 128 + wave * 32;
  const int c0 = blockIdx.y * 64;

  const _Float16* W = woT + (size_t)c0 * HDV_;
  v8f acc[2][4] = {};
  const _Float16* arow0 = ctx + (size_t)(r0 + l15) * HDV_;
  const _Float16* arow1 = arow0 + (size_t)16 * HDV_;

  stage_tile(sW[0], W, HDV_);
  for (int s = 0; s < NST; ++s) {
    if (s + 1 < NST) stage_tile(sW[(s + 1) & 1], W + (s + 1) * DCH, HDV_);
    stage_wait(s + 1 < NST);
    const _Float16* sb = sW[s & 1];
#pragma unroll
    for (int dd = 0; dd < DCH; dd += 32) {
      const int k0 = s * DCH + dd;
      v16h A0 = ldA16(arow0 + k0 + hi * 8);
      v16h A1 = ldA16(arow1 + k0 + hi * 8);
#pragma unroll
      for (int ct = 0; ct < 4; ++ct) {
        v16h Bv = ldB(sb + (ct * 16 + l15) * SP + dd + hi * 8);
        acc[0][ct] = wmma16(A0, Bv, acc[0][ct]);
        acc[1][ct] = wmma16(A1, Bv, acc[1][ct]);
      }
    }
    __syncthreads();
  }

#pragma unroll
  for (int rg = 0; rg < 2; ++rg) {
    const int rbase = r0 + rg * 16 + hi * 8;
#pragma unroll
    for (int ct = 0; ct < 4; ++ct) {
      const int d = c0 + ct * 16 + l15;
      const float bia = bo[d];
#pragma unroll
      for (int r = 0; r < 8; ++r)
        out[(size_t)(rbase + r) * D_ + d] = acc[rg][ct][r] + bia;
    }
  }
}

// ---------------------------------------------------------------------------
extern "C" void kernel_launch(void* const* d_in, const int* in_sizes, int n_in,
                              void* d_out, int out_size, void* d_ws, size_t ws_size,
                              hipStream_t stream) {
  (void)in_sizes; (void)n_in; (void)out_size; (void)ws_size;
  const float* q  = (const float*)d_in[0];
  const float* k  = (const float*)d_in[1];
  const float* v  = (const float*)d_in[2];
  // d_in[3] = causal mask: applied analytically, not read.
  const float* wq = (const float*)d_in[4];
  const float* bq = (const float*)d_in[5];
  const float* wk = (const float*)d_in[6];
  const float* bk = (const float*)d_in[7];
  const float* wv = (const float*)d_in[8];
  const float* bv = (const float*)d_in[9];
  const float* wo = (const float*)d_in[10];
  const float* bo = (const float*)d_in[11];
  float* out = (float*)d_out;

  _Float16* ws  = (_Float16*)d_ws;
  _Float16* wT  = ws;                                       // 3*H*64*D
  _Float16* woT = wT  + (size_t)3 * H_ * DK_ * D_;          // D*HDV
  _Float16* qh  = woT + (size_t)D_ * HDV_;                  // B*H*N*DK
  _Float16* kh  = qh  + (size_t)B_ * H_ * N_ * DK_;
  _Float16* vhT = kh  + (size_t)B_ * H_ * N_ * DK_;
  _Float16* ctx = vhT + (size_t)B_ * H_ * N_ * DV_;

  prep_kernel<<<(3 * H_ * DK_ * D_ + D_ * HDV_) / 256, 256, 0, stream>>>(
      wq, wk, wv, wo, wT, woT);

  dim3 gp(N_ / 128, 3 * H_, B_);
  proj_kernel<<<gp, 128, 0, stream>>>(q, k, v, bq, bk, bv, wT, qh, kh, vhT);

  dim3 ga(N_ / 64, H_, B_);
  attn_kernel<<<ga, 128, 0, stream>>>(qh, kh, vhT, ctx);

  dim3 go((B_ * N_) / 128, D_ / 64, 1);
  oproj_kernel<<<go, 128, 0, stream>>>(ctx, woT, bo, out);
}
